// CustomReservoir_56014963475183
// MI455X (gfx1250) — matrix operand order repaired and checked by reading, not compiled, not run on hardware
//
#include <hip/hip_runtime.h>
#include <stdint.h>

#define SEQ_LEN 8192
#define IN_SIZE 512
#define N_RES   2048
#define NB      64            // persistent blocks in the scan kernel
#define RPB     (N_RES / NB)  // rows of W_res owned per block = 32
#define TDM_ROWS 8            // rows per TDM chunk (8*2048*4B = 64 KB staging)

typedef __attribute__((ext_vector_type(16))) __bf16 v16bf;
typedef __attribute__((ext_vector_type(8)))  float  v8f;
typedef unsigned int u32x4 __attribute__((ext_vector_type(4)));
typedef int          i32x4 __attribute__((ext_vector_type(4)));
typedef int          i32x8 __attribute__((ext_vector_type(8)));

union ABReg { uint32_t u[8]; v16bf v; };

__device__ __forceinline__ uint32_t pk_bf16(float lo, float hi) {
    union { __bf16 b[2]; uint32_t u; } t;
    t.b[0] = (__bf16)lo; t.b[1] = (__bf16)hi;
    return t.u;
}

// ---------------------------------------------------------------------------
// TDM: DMA a 2-D fp32 tile (rows x width) from global memory into LDS.
// Descriptor built per CDNA5 ISA ch.8 (D# groups 0/1; groups 2/3 zero = 2-D).
// ---------------------------------------------------------------------------
__device__ __forceinline__ void tdm_load_2d_f32(const void* gaddr, uint32_t lds_off,
                                                uint32_t width, uint32_t rows,
                                                uint32_t row_stride) {
    unsigned long long ga = (unsigned long long)(uintptr_t)gaddr;
    u32x4 g0;
    g0[0] = 1u;                                             // count=1, user mode
    g0[1] = lds_off;                                        // lds_addr
    g0[2] = (uint32_t)ga;                                   // global_addr[31:0]
    g0[3] = (uint32_t)((ga >> 32) & 0x01FFFFFFull)          // global_addr[56:32]
          | (2u << 30);                                     // type = 2 ("image")
    i32x8 g1;
    g1[0] = (int)(2u << 16);                                // wg_mask=0, data_size=2 (4B)
    g1[1] = (int)((width & 0xFFFFu) << 16);                 // tensor_dim0[15:0] @63:48
    g1[2] = (int)((width >> 16) & 0xFFFFu)                  // tensor_dim0[31:16] @79:64
          | (int)((rows & 0xFFFFu) << 16);                  // tensor_dim1[15:0] @95:80
    g1[3] = (int)((rows >> 16) & 0xFFFFu)                   // tensor_dim1[31:16] @111:96
          | (int)((width & 0xFFFFu) << 16);                 // tile_dim0 @127:112
    g1[4] = (int)(rows & 0xFFFFu);                          // tile_dim1 @143:128, tile_dim2=0
    g1[5] = (int)row_stride;                                // tensor_dim0_stride[31:0]
    g1[6] = 0;                                              // stride hi / dim1_stride lo
    g1[7] = 0;
    i32x4 z4 = {0, 0, 0, 0};
#if defined(__clang_major__) && __clang_major__ >= 23
    i32x8 z8 = {0, 0, 0, 0, 0, 0, 0, 0};
    __builtin_amdgcn_tensor_load_to_lds(g0, g1, z4, z4, z8, 0);
#else
    __builtin_amdgcn_tensor_load_to_lds(g0, g1, z4, z4, 0);
#endif
}

// ---------------------------------------------------------------------------
// device-wide barrier for the persistent scan kernel (gen-counter pattern)
// sync[0] = arrival counter, sync[16] = generation
// ---------------------------------------------------------------------------
__device__ __forceinline__ void grid_barrier(int* sync) {
    __syncthreads();
    __threadfence();   // make this block's global stores visible device-wide
    if (threadIdx.x == 0) {
        int* cnt = sync;
        int* gen = sync + 16;
        int g   = __hip_atomic_load(gen, __ATOMIC_ACQUIRE, __HIP_MEMORY_SCOPE_AGENT);
        int old = __hip_atomic_fetch_add(cnt, 1, __ATOMIC_ACQ_REL, __HIP_MEMORY_SCOPE_AGENT);
        if (old == NB - 1) {
            __hip_atomic_store(cnt, 0, __ATOMIC_RELAXED, __HIP_MEMORY_SCOPE_AGENT);
            __hip_atomic_fetch_add(gen, 1, __ATOMIC_ACQ_REL, __HIP_MEMORY_SCOPE_AGENT);
        } else {
            while (__hip_atomic_load(gen, __ATOMIC_ACQUIRE, __HIP_MEMORY_SCOPE_AGENT) == g)
                __builtin_amdgcn_s_sleep(2);
        }
    }
    __syncthreads();
}

__global__ void resv_init(int* sync) {
    if (threadIdx.x < 32) sync[threadIdx.x] = 0;
}

// ---------------------------------------------------------------------------
// Phase 1: u[t, r] = sum_k input[t,k] * W_in[r,k]   (8192 x 2048 x 512 GEMM)
// One 16x64 output tile per wave: A fragment reused across 4 WMMAs per K-step.
// ---------------------------------------------------------------------------
__global__ __launch_bounds__(256)
void resv_gemm_uin(const float* __restrict__ input, const float* __restrict__ W_in,
                   float* __restrict__ u) {
    const int wid  = (blockIdx.x * 256 + threadIdx.x) >> 5;
    const int lane = threadIdx.x & 31;
    const int l    = lane & 15;
    const int half = lane >> 4;
    const int ti   = wid >> 5;          // 0..511 : row tile of u
    const int tj0  = (wid & 31) * 4;    // first of 4 col tiles

    const float* arow = input + (ti * 16 + l) * IN_SIZE;    // A: M = l (both halves)
    const float* brow[4];
#pragma unroll
    for (int q = 0; q < 4; ++q)
        brow[q] = W_in + ((tj0 + q) * 16 + l) * IN_SIZE;    // B col n=l is W_in row n

    v8f c[4] = {};
    for (int kb = 0; kb < IN_SIZE; kb += 32) {
        ABReg a;
#pragma unroll
        for (int j = 0; j < 4; ++j) {                       // A 16x32 bf16 layout
            int ka = kb + half * 8 + 2 * j;
            a.u[j]     = pk_bf16(arow[ka],      arow[ka + 1]);
            a.u[j + 4] = pk_bf16(arow[ka + 16], arow[ka + 17]);
        }
#pragma unroll
        for (int q = 0; q < 4; ++q) {
            ABReg b;
#pragma unroll
            for (int j = 0; j < 8; ++j) {                   // B 32x16 bf16 layout
                int kk = kb + half * 8 + 2 * j + (j >= 4 ? 8 : 0);
                b.u[j] = pk_bf16(brow[q][kk], brow[q][kk + 1]);
            }
            c[q] = __builtin_amdgcn_wmma_f32_16x16x32_bf16(false, a.v, false, b.v,
                                                           (short)0, c[q], false, false);
        }
    }
    // D: VGPR r, lanes0-15 -> (M=r, N=lane); lanes16-31 -> (M=8+r, N=lane-16)
#pragma unroll
    for (int q = 0; q < 4; ++q) {
        float* o = u + (size_t)(ti * 16 + half * 8) * N_RES + (tj0 + q) * 16 + l;
#pragma unroll
        for (int r = 0; r < 8; ++r) o[r * N_RES] = c[q][r];
    }
}

// ---------------------------------------------------------------------------
// Phase 2: persistent scan. Each of 64 blocks pins 32 rows of W_res in LDS as
// bf16 (128 KB), filled at startup via TDM 2-D tile DMA (tensor_load_to_lds +
// s_wait_tensorcnt). Per step: the 8 KB state row arrives by
// global_load_async_to_lds_b128 (ASYNCcnt), is converted to bf16, and the GEMV
// runs on the WMMA pipe with s broadcast across B's 16 columns.
// ---------------------------------------------------------------------------
__global__ __launch_bounds__(256)
void resv_scan(const float* __restrict__ state0,
               const float* __restrict__ W_res,
               const float* __restrict__ u,
               float* __restrict__ out,
               int* __restrict__ sync) {
    extern __shared__ char smem[];
    // LDS map (bytes):
    //   [0,           131072) W bf16   32 x 2048
    //   [131072,      135168) s bf16   2048
    //   [135168,      143360) s fp32 staging (async dst), 8 KB
    //   [143360,      208896) W fp32 staging (TDM dst),  64 KB
    //   [208896,      209024) y partials, 32 f32
    __bf16* Wl     = (__bf16*)smem;
    __bf16* sl     = (__bf16*)(smem + 131072);
    float*  sstage = (float*) (smem + 135168);
    float*  wstage = (float*) (smem + 143360);
    float*  yl     = (float*) (smem + 208896);

    const int tid  = threadIdx.x;
    const int blk  = blockIdx.x;
    const int lane = tid & 31;
    const int l    = lane & 15;
    const int half = lane >> 4;
    const int wave = tid >> 5;   // 0..7
    const int rt   = wave >> 2;  // row tile within block: 0..1
    const int kq   = wave & 3;   // K quarter: 0..3
    const int rowbase = blk * RPB;

    // ---- TDM fill: 4 chunks of 8 rows, DMA fp32 -> staging, convert to bf16
    const uint32_t wstage_lds = (uint32_t)(uintptr_t)wstage;
    const int wave_u = __builtin_amdgcn_readfirstlane(wave);  // force uniform branch
    for (int ch = 0; ch < RPB / TDM_ROWS; ++ch) {
        if (wave_u == 0) {
            tdm_load_2d_f32(W_res + (size_t)(rowbase + ch * TDM_ROWS) * N_RES,
                            wstage_lds, N_RES, TDM_ROWS, N_RES);
            __builtin_amdgcn_s_wait_tensorcnt(0);
        }
        __syncthreads();
        for (int i = tid; i < TDM_ROWS * N_RES; i += 256)
            Wl[ch * TDM_ROWS * N_RES + i] = (__bf16)wstage[i];
        __syncthreads();
    }

    // block 0 publishes the initial state as output row 0
    if (blk == 0)
        for (int i = tid; i < N_RES; i += 256)
            out[i] = state0[i];

    grid_barrier(sync);  // row 0 visible everywhere before step 0

    const float inv_sqrt_n = 0.02209708691207961f; // 1/sqrt(2048)
    const uint32_t stage_lds = (uint32_t)(uintptr_t)sstage;

    for (int t = 0; t < SEQ_LEN; ++t) {
        const float* s = out + (size_t)t * N_RES;

        // async DMA the 8 KB state row into LDS staging: 512 x b128 chunks
#pragma unroll
        for (int ci = 0; ci < 2; ++ci) {
            int idx = tid + 256 * ci;                             // chunk index 0..511
            unsigned long long ga = (unsigned long long)(uintptr_t)s + (unsigned)idx * 16u;
            uint32_t la = stage_lds + (unsigned)idx * 16u;
            asm volatile("global_load_async_to_lds_b128 %0, %1, off"
                         :: "v"(la), "v"(ga) : "memory");
        }
        if (tid < RPB) yl[tid] = 0.0f;
        asm volatile("s_wait_asynccnt 0x0" ::: "memory");
        __syncthreads();

        // convert staged fp32 -> bf16 operand vector
        for (int i = tid; i < N_RES; i += 256)
            sl[i] = (__bf16)sstage[i];
        __syncthreads();

        // each wave: one 16-row tile, one 512-wide K quarter (16 WMMAs)
        v8f c = {};
        const int k0 = kq * (N_RES / 4);
        const __bf16* wr = &Wl[(rt * 16 + l) * N_RES];
        for (int kb = k0; kb < k0 + N_RES / 4; kb += 32) {
            ABReg a, b;
#pragma unroll
            for (int j = 0; j < 4; ++j) {
                int ka = kb + half * 8 + 2 * j;
                a.u[j]     = *(const uint32_t*)&wr[ka];
                a.u[j + 4] = *(const uint32_t*)&wr[ka + 16];
            }
#pragma unroll
            for (int j = 0; j < 8; ++j)     // B = s broadcast into all 16 columns
                b.u[j] = *(const uint32_t*)&sl[kb + half * 8 + 2 * j + (j >= 4 ? 8 : 0)];
            c = __builtin_amdgcn_wmma_f32_16x16x32_bf16(false, a.v, false, b.v,
                                                        (short)0, c, false, false);
        }
        // column 0 of D lives in lanes 0 (M=0..7) and 16 (M=8..15)
        if (l == 0) {
#pragma unroll
            for (int r = 0; r < 8; ++r)
                atomicAdd(&yl[rt * 16 + half * 8 + r], c[r]);
        }
        __syncthreads();

        if (tid < RPB) {
            int r = rowbase + tid;
            float sn = tanhf(u[(size_t)t * N_RES + r] + yl[tid]) * inv_sqrt_n;
            out[(size_t)(t + 1) * N_RES + r] = sn;
        }
        grid_barrier(sync);  // row t+1 visible before any block reads it
    }
}

// ---------------------------------------------------------------------------
extern "C" void kernel_launch(void* const* d_in, const int* in_sizes, int n_in,
                              void* d_out, int out_size, void* d_ws, size_t ws_size,
                              hipStream_t stream) {
    const float* input = (const float*)d_in[0];   // (8192, 512)
    const float* state = (const float*)d_in[1];   // (2048,)
    const float* W_in  = (const float*)d_in[2];   // (2048, 512)
    const float* W_res = (const float*)d_in[3];   // (2048, 2048)
    float* out  = (float*)d_out;                  // (8193, 2048)
    int*   sync = (int*)d_ws;                     // first 256 B: barrier state
    float* u    = (float*)((char*)d_ws + 256);    // 8192*2048 fp32 = 64 MB scratch

    resv_init<<<1, 64, 0, stream>>>(sync);
    // 512 row tiles x 32 col-tile groups = 16384 waves = 2048 blocks
    resv_gemm_uin<<<2048, 256, 0, stream>>>(input, W_in, u);

    const size_t smem = 209024;  // see LDS map in resv_scan
    resv_scan<<<NB, 256, smem, stream>>>(state, W_res, u, out, sync);
}
